// DecoderCoords_45904610460123
// MI455X (gfx1250) — matrix operand order repaired and checked by reading, not compiled, not run on hardware
//
#include <hip/hip_runtime.h>
#include <hip/hip_bf16.h>

typedef __bf16 bf16;
typedef __attribute__((ext_vector_type(16))) __bf16 bf16x16;
typedef __attribute__((ext_vector_type(8)))  __bf16 bf16x8;
typedef __attribute__((ext_vector_type(8)))  float  f32x8;
typedef __attribute__((ext_vector_type(4)))  float  f32x4;

#define B_  8
#define T_  64
#define A_  16
#define D_  512
#define H_  8
#define HD_ 64
#define F_  2048
#define L_  6
#define M_  1024
#define S_  (T_*A_)    /* 1024 */
#define NX_ (B_*S_)    /* 8192 */
#define NM_ (B_*M_)    /* 8192 */

__device__ __forceinline__ f32x8 wmma_bf16f32(bf16x16 a, bf16x16 b, f32x8 c) {
  // v_wmma_f32_16x16x32_bf16  (neg_a, A, neg_b, B, c_mod, C, reuse_a, reuse_b)
  return __builtin_amdgcn_wmma_f32_16x16x32_bf16(false, a, false, b, (short)0, c, false, false);
}

__device__ __forceinline__ bf16x16 cat8(bf16x8 lo, bf16x8 hi) {
  return __builtin_shufflevector(lo, hi, 0, 1, 2, 3, 4, 5, 6, 7,
                                 8, 9, 10, 11, 12, 13, 14, 15);
}

__device__ __forceinline__ float rmax16(float v) {
#pragma unroll
  for (int m = 8; m >= 1; m >>= 1) v = fmaxf(v, __shfl_xor(v, m, 32));
  return v;
}
__device__ __forceinline__ float rsum16(float v) {
#pragma unroll
  for (int m = 8; m >= 1; m >>= 1) v += __shfl_xor(v, m, 32);
  return v;
}

// ---------------------------------------------------------------------------
// Elementwise helpers
// ---------------------------------------------------------------------------
__global__ void add3_kernel(const float* __restrict__ a, const float* __restrict__ b,
                            const float* __restrict__ c, float* __restrict__ out, int n) {
  int i = blockIdx.x * blockDim.x + threadIdx.x;
  if (i < n) out[i] = a[i] + b[i] + c[i];
}

__global__ void f32_to_bf16_kernel(const float* __restrict__ in, bf16* __restrict__ out, int n) {
  int i = blockIdx.x * blockDim.x + threadIdx.x;
  if (i < n) out[i] = (bf16)in[i];
}

// ---------------------------------------------------------------------------
// LayerNorm: one 256-thread block per row of length D
// ---------------------------------------------------------------------------
__global__ __launch_bounds__(256) void ln_kernel(
    const float* __restrict__ in, const float* __restrict__ g,
    const float* __restrict__ bb, float* __restrict__ outf,
    bf16* __restrict__ outb, int D)
{
  __shared__ float red[16];
  const int row = blockIdx.x;
  const float* x = in + (size_t)row * D;
  float s = 0.f, s2 = 0.f;
  for (int i = threadIdx.x; i < D; i += blockDim.x) {
    float v = x[i]; s += v; s2 += v * v;
  }
#pragma unroll
  for (int m = 16; m >= 1; m >>= 1) { s += __shfl_xor(s, m, 32); s2 += __shfl_xor(s2, m, 32); }
  const int wave = threadIdx.x >> 5, lane = threadIdx.x & 31;
  if (lane == 0) { red[wave] = s; red[8 + wave] = s2; }
  __syncthreads();
  if (threadIdx.x == 0) {
    float ts = 0.f, ts2 = 0.f;
    for (int w = 0; w < (int)(blockDim.x >> 5); ++w) { ts += red[w]; ts2 += red[8 + w]; }
    red[0] = ts; red[8] = ts2;
  }
  __syncthreads();
  const float mean = red[0] / D;
  const float var  = red[8] / D - mean * mean;
  const float rstd = rsqrtf(var + 1e-5f);
  for (int i = threadIdx.x; i < D; i += blockDim.x) {
    float y = (x[i] - mean) * rstd * g[i] + bb[i];
    if (outf) outf[(size_t)row * D + i] = y;
    if (outb) outb[(size_t)row * D + i] = (bf16)y;
  }
}

// ---------------------------------------------------------------------------
// GEMM: Y[N,Mo] = X[N,K](bf16) @ W[Mo,K](f32->bf16)^T + bias (+res) (opt ReLU)
// Block = 256 threads = 8 waves; each wave = 16 rows x 32 cols (2 WMMA accs).
// Branchless inner loop: b128 loads only. Out-of-range output columns read
// clamped row 0 of W (in-bounds); their D columns are never stored, and WMMA
// columns are independent, so no zero-fill is needed. The epilogue has a
// wave-uniform fast path for full tiles (all call sites but the Mo=4 head).
// ---------------------------------------------------------------------------
__global__ __launch_bounds__(256) void gemm_bf16_kernel(
    const bf16* __restrict__ X, int ldx,
    const float* __restrict__ W, int ldw,
    const float* __restrict__ bias,
    const float* __restrict__ residual,
    float* __restrict__ Yf, bf16* __restrict__ Yb,
    int N, int K, int Mo, int relu)
{
  const int wave = threadIdx.x >> 5;
  const int lane = threadIdx.x & 31;
  const int half = lane >> 4;
  const int l15  = lane & 15;
  const int r0 = blockIdx.x * 64 + (wave & 3) * 16;
  const int c0 = blockIdx.y * 64 + (wave >> 2) * 32;
  f32x8 acc0 = {}; f32x8 acc1 = {};
  const bf16* xrow = X + (size_t)(r0 + l15) * ldx;   // A row m = lane&15
  const int ca = c0 + l15;
  const int cb = c0 + 16 + l15;
  const bool va = ca < Mo;
  const bool vb = cb < Mo;
  const float* wrow0 = W + (size_t)(va ? ca : 0) * ldw;
  const float* wrow1 = W + (size_t)(vb ? cb : 0) * ldw;
#pragma unroll 2
  for (int kk = 0; kk < K; kk += 32) {
    // A: lanes<16 hold K 0-15; lanes>=16 hold K 16-31
    const bf16x8* xq = (const bf16x8*)(xrow + kk + half * 16);
    bf16x16 a = cat8(xq[0], xq[1]);
    // B: lanes<16 hold K 0-15; lanes>=16 hold K 16-31 (element order = K order)
    const f32x4* wq0 = (const f32x4*)(wrow0 + kk + half * 16);
    const f32x4* wq1 = (const f32x4*)(wrow1 + kk + half * 16);
    f32x4 w00 = wq0[0], w01 = wq0[1], w02 = wq0[2], w03 = wq0[3];
    f32x4 w10 = wq1[0], w11 = wq1[1], w12 = wq1[2], w13 = wq1[3];
    bf16x16 b0, b1;
#pragma unroll
    for (int e = 0; e < 4; ++e) {
      b0[e]      = (bf16)w00[e]; b0[4 + e]  = (bf16)w01[e];
      b0[8 + e]  = (bf16)w02[e]; b0[12 + e] = (bf16)w03[e];
      b1[e]      = (bf16)w10[e]; b1[4 + e]  = (bf16)w11[e];
      b1[8 + e]  = (bf16)w12[e]; b1[12 + e] = (bf16)w13[e];
    }
    acc0 = wmma_bf16f32(a, b0, acc0);
    acc1 = wmma_bf16f32(a, b1, acc1);
  }
  if ((c0 + 32) <= Mo) {
    // Fast path: full tile, no lane divergence (all pointer checks uniform).
    const float bia0 = bias ? bias[ca] : 0.f;
    const float bia1 = bias ? bias[cb] : 0.f;
#pragma unroll
    for (int v = 0; v < 8; ++v) {
      const int row = r0 + v + half * 8;        // D layout: M = v + 8*(lane>=16)
      float y0 = acc0[v] + bia0;
      float y1 = acc1[v] + bia1;
      if (residual) { y0 += residual[(size_t)row * Mo + ca];
                      y1 += residual[(size_t)row * Mo + cb]; }
      if (relu) { y0 = fmaxf(y0, 0.f); y1 = fmaxf(y1, 0.f); }
      if (Yf) { Yf[(size_t)row * Mo + ca] = y0; Yf[(size_t)row * Mo + cb] = y1; }
      if (Yb) { Yb[(size_t)row * Mo + ca] = (bf16)y0; Yb[(size_t)row * Mo + cb] = (bf16)y1; }
    }
  } else {
    // Ragged tile (head GEMM, Mo=4): per-lane guards.
#pragma unroll
    for (int v = 0; v < 8; ++v) {
      const int row = r0 + v + half * 8;
      float y0 = acc0[v];
      float y1 = acc1[v];
      if (bias)     { if (va) y0 += bias[ca]; if (vb) y1 += bias[cb]; }
      if (residual) { if (va) y0 += residual[(size_t)row * Mo + ca];
                      if (vb) y1 += residual[(size_t)row * Mo + cb]; }
      if (relu) { y0 = fmaxf(y0, 0.f); y1 = fmaxf(y1, 0.f); }
      if (va) { if (Yf) Yf[(size_t)row * Mo + ca] = y0; if (Yb) Yb[(size_t)row * Mo + ca] = (bf16)y0; }
      if (vb) { if (Yf) Yf[(size_t)row * Mo + cb] = y1; if (Yb) Yb[(size_t)row * Mo + cb] = (bf16)y1; }
    }
  }
}

// ---------------------------------------------------------------------------
// Flash attention: 1 wave / 16-query tile / head. HD=64.
// Scores via 4 WMMAs per 32-key block, online softmax, P->A via LDS, P@V WMMAs.
// Block-causal mask: tile rows share one timestep -> key < q0+16.
// All K/V reads are in-bounds: ceil32(limit) <= Skv for every call site.
// ---------------------------------------------------------------------------
__global__ __launch_bounds__(32) void attn_kernel(
    const bf16* __restrict__ Q, int ldq,
    const bf16* __restrict__ Km, int ldk,
    const bf16* __restrict__ Vm, int ldv,
    bf16* __restrict__ O, int ldo,
    int Sq, int Skv, int causal)
{
  __shared__ float plds[16][33];
  const int lane = threadIdx.x;
  const int half = lane >> 4;
  const int l15  = lane & 15;
  const int q0 = blockIdx.x * 16;
  const int h  = blockIdx.y;
  const int b  = blockIdx.z;
  const bf16* Qb = Q  + ((size_t)b * Sq  + q0) * ldq + h * HD_;
  const bf16* Kb = Km +  (size_t)b * Skv * ldk + h * HD_;
  const bf16* Vb = Vm +  (size_t)b * Skv * ldv + h * HD_;

  bf16x16 aq0, aq1;  // Q tile 16x64 as two A fragments (K 0-31, 32-63)
  {
    const bf16x8* qp = (const bf16x8*)(Qb + (size_t)l15 * ldq + half * 16);
    aq0 = cat8(qp[0], qp[1]);   // K 0-15 / 16-31
    aq1 = cat8(qp[4], qp[5]);   // K 32-47 / 48-63
  }
  f32x8 o[4];
  f32x8 zero = {};
#pragma unroll
  for (int nt = 0; nt < 4; ++nt) o[nt] = zero;
  float mrun[8], lrun[8];
#pragma unroll
  for (int v = 0; v < 8; ++v) { mrun[v] = -1e30f; lrun[v] = 0.f; }

  const int limit = causal ? (q0 + 16) : Skv;
  const int nkb = (limit + 31) >> 5;
  for (int kb = 0; kb < nkb; ++kb) {
    const int k0 = kb * 32;
    // K^T fragments: 2 key tiles x 2 K-steps; element order = hd-dim order
    const bf16x8* kp0 = (const bf16x8*)(Kb + (size_t)(k0 + l15) * ldk + half * 16);
    const bf16x8* kp1 = (const bf16x8*)(Kb + (size_t)(k0 + 16 + l15) * ldk + half * 16);
    bf16x16 b00 = cat8(kp0[0], kp0[1]);
    bf16x16 b01 = cat8(kp0[4], kp0[5]);
    bf16x16 b10 = cat8(kp1[0], kp1[1]);
    bf16x16 b11 = cat8(kp1[4], kp1[5]);

    f32x8 s0 = {}; f32x8 s1 = {};
    s0 = wmma_bf16f32(aq0, b00, s0);
    s0 = wmma_bf16f32(aq1, b01, s0);
    s1 = wmma_bf16f32(aq0, b10, s1);
    s1 = wmma_bf16f32(aq1, b11, s1);

    const bool in0 = (k0 + l15) < limit;
    const bool in1 = (k0 + 16 + l15) < limit;
#pragma unroll
    for (int v = 0; v < 8; ++v) {
      float x0 = in0 ? s0[v] * 0.125f : -1e30f;   // 1/sqrt(HD)=0.125
      float x1 = in1 ? s1[v] * 0.125f : -1e30f;
      float mnew  = fmaxf(mrun[v], rmax16(fmaxf(x0, x1)));
      float alpha = __expf(mrun[v] - mnew);
      mrun[v] = mnew;
      float p0 = __expf(x0 - mnew);
      float p1 = __expf(x1 - mnew);
      lrun[v] = lrun[v] * alpha + rsum16(p0 + p1);
#pragma unroll
      for (int nt = 0; nt < 4; ++nt) o[nt][v] *= alpha;
      plds[v + half * 8][l15]      = p0;   // C-layout -> row-major P in LDS
      plds[v + half * 8][16 + l15] = p1;
    }
    __syncthreads();
    bf16x16 pa;  // P as A fragment (16x32)
    {
      const float* pr = &plds[l15][0];
#pragma unroll
      for (int e = 0; e < 8; ++e) {
        pa[e]     = (bf16)pr[half * 16 + e];
        pa[8 + e] = (bf16)pr[half * 16 + 8 + e];
      }
    }
#pragma unroll
    for (int nt = 0; nt < 4; ++nt) {
      bf16x16 bv;  // V block 32xHD, N tile nt (strided column gather)
#pragma unroll
      for (int e = 0; e < 16; ++e)
        bv[e] = Vb[(size_t)(k0 + half * 16 + e) * ldv + nt * 16 + l15];
      o[nt] = wmma_bf16f32(pa, bv, o[nt]);
    }
    __syncthreads();
  }
  bf16* Ob = O + ((size_t)b * Sq + q0) * ldo + h * HD_;
#pragma unroll
  for (int v = 0; v < 8; ++v) {
    const float inv = 1.0f / lrun[v];
    const int row = v + half * 8;
#pragma unroll
    for (int nt = 0; nt < 4; ++nt)
      Ob[(size_t)row * ldo + nt * 16 + l15] = (bf16)(o[nt][v] * inv);
  }
}

// ---------------------------------------------------------------------------
// Host orchestration
// ---------------------------------------------------------------------------
extern "C" void kernel_launch(void* const* d_in, const int* in_sizes, int n_in,
                              void* d_out, int out_size, void* d_ws, size_t ws_size,
                              hipStream_t stream) {
  (void)in_sizes; (void)n_in; (void)out_size; (void)ws_size;
  const float* coords   = (const float*)d_in[0];
  const float* ids      = (const float*)d_in[1];
  const float* tsteps   = (const float*)d_in[2];
  const float* enc      = (const float*)d_in[3];
  const float* ln_emb_g = (const float*)d_in[4];
  const float* ln_emb_b = (const float*)d_in[5];
  const float* sa_qkv_w = (const float*)d_in[6];
  const float* sa_qkv_b = (const float*)d_in[7];
  const float* sa_out_w = (const float*)d_in[8];
  const float* sa_out_b = (const float*)d_in[9];
  const float* ca_qkv_w = (const float*)d_in[10];
  const float* ca_qkv_b = (const float*)d_in[11];
  const float* ca_out_w = (const float*)d_in[12];
  const float* ca_out_b = (const float*)d_in[13];
  const float* ffn_w1   = (const float*)d_in[14];
  const float* ffn_b1   = (const float*)d_in[15];
  const float* ffn_w2   = (const float*)d_in[16];
  const float* ffn_b2   = (const float*)d_in[17];
  const float* ln1_g    = (const float*)d_in[18];
  const float* ln1_b    = (const float*)d_in[19];
  const float* ln2_g    = (const float*)d_in[20];
  const float* ln2_b    = (const float*)d_in[21];
  const float* ln3_g    = (const float*)d_in[22];
  const float* ln3_b    = (const float*)d_in[23];
  const float* mlp_w1   = (const float*)d_in[24];
  const float* mlp_b1   = (const float*)d_in[25];
  const float* mlp_w2   = (const float*)d_in[26];
  const float* mlp_b2   = (const float*)d_in[27];

  char* ws = (char*)d_ws;
  size_t off = 0;
  auto alloc = [&](size_t bytes) -> char* {
    char* p = ws + off;
    off += (bytes + 255) & ~(size_t)255;
    return p;
  };
  float* xf   = (float*)alloc((size_t)NX_ * D_ * 4);       // current activations f32
  float* resf = (float*)alloc((size_t)NX_ * D_ * 4);       // pre-LN sum
  bf16*  xb   = (bf16*) alloc((size_t)NX_ * D_ * 2);       // current activations bf16
  bf16*  memb = (bf16*) alloc((size_t)NM_ * D_ * 2);       // encoder memory bf16
  bf16*  qkvb = (bf16*) alloc((size_t)NX_ * (3*D_) * 2);   // self-attn QKV
  bf16*  attnb= (bf16*) alloc((size_t)NX_ * D_ * 2);       // attention output
  bf16*  qcab = (bf16*) alloc((size_t)NX_ * D_ * 2);       // cross-attn Q
  bf16*  kvcab= (bf16*) alloc((size_t)NM_ * (2*D_) * 2);   // cross-attn K,V
  bf16*  h1b  = (bf16*) alloc((size_t)NX_ * F_ * 2);       // FFN hidden
  bf16*  hb   = (bf16*) alloc((size_t)NX_ * D_ * 2);       // head hidden

  const int nelem = NX_ * D_;
  add3_kernel<<<(nelem + 255) / 256, 256, 0, stream>>>(coords, ids, tsteps, resf, nelem);
  ln_kernel<<<NX_, 256, 0, stream>>>(resf, ln_emb_g, ln_emb_b, xf, xb, D_);
  f32_to_bf16_kernel<<<(NM_ * D_ + 255) / 256, 256, 0, stream>>>(enc, memb, NM_ * D_);

  auto gemm = [&](const bf16* X, int ldx, const float* W, int ldw,
                  const float* bias, const float* residual,
                  float* Yf, bf16* Yb, int N, int K, int Mo, int relu) {
    dim3 grid((N + 63) / 64, (Mo + 63) / 64);
    gemm_bf16_kernel<<<grid, 256, 0, stream>>>(X, ldx, W, ldw, bias, residual,
                                               Yf, Yb, N, K, Mo, relu);
  };

  for (int l = 0; l < L_; ++l) {
    // --- self attention ---
    gemm(xb, D_, sa_qkv_w + (size_t)l * 3 * D_ * D_, D_, sa_qkv_b + (size_t)l * 3 * D_,
         nullptr, nullptr, qkvb, NX_, D_, 3 * D_, 0);
    {
      dim3 grid(S_ / 16, H_, B_);
      attn_kernel<<<grid, 32, 0, stream>>>(qkvb, 3 * D_, qkvb + D_, 3 * D_,
                                           qkvb + 2 * D_, 3 * D_,
                                           attnb, D_, S_, S_, 1);
    }
    gemm(attnb, D_, sa_out_w + (size_t)l * D_ * D_, D_, sa_out_b + (size_t)l * D_,
         xf, resf, nullptr, NX_, D_, D_, 0);
    ln_kernel<<<NX_, 256, 0, stream>>>(resf, ln1_g + (size_t)l * D_, ln1_b + (size_t)l * D_,
                                       xf, xb, D_);
    // --- cross attention ---
    gemm(xb, D_, ca_qkv_w + (size_t)l * 3 * D_ * D_, D_, ca_qkv_b + (size_t)l * 3 * D_,
         nullptr, nullptr, qcab, NX_, D_, D_, 0);                       // Wq rows
    gemm(memb, D_, ca_qkv_w + (size_t)l * 3 * D_ * D_ + (size_t)D_ * D_, D_,
         ca_qkv_b + (size_t)l * 3 * D_ + D_,
         nullptr, nullptr, kvcab, NM_, D_, 2 * D_, 0);                  // Wk,Wv rows
    {
      dim3 grid(S_ / 16, H_, B_);
      attn_kernel<<<grid, 32, 0, stream>>>(qcab, D_, kvcab, 2 * D_, kvcab + D_, 2 * D_,
                                           attnb, D_, S_, M_, 0);
    }
    gemm(attnb, D_, ca_out_w + (size_t)l * D_ * D_, D_, ca_out_b + (size_t)l * D_,
         xf, resf, nullptr, NX_, D_, D_, 0);
    ln_kernel<<<NX_, 256, 0, stream>>>(resf, ln2_g + (size_t)l * D_, ln2_b + (size_t)l * D_,
                                       xf, xb, D_);
    // --- FFN ---
    gemm(xb, D_, ffn_w1 + (size_t)l * F_ * D_, D_, ffn_b1 + (size_t)l * F_,
         nullptr, nullptr, h1b, NX_, D_, F_, 1);
    gemm(h1b, F_, ffn_w2 + (size_t)l * D_ * F_, F_, ffn_b2 + (size_t)l * D_,
         xf, resf, nullptr, NX_, F_, D_, 0);
    ln_kernel<<<NX_, 256, 0, stream>>>(resf, ln3_g + (size_t)l * D_, ln3_b + (size_t)l * D_,
                                       xf, xb, D_);
  }
  // --- head MLP ---
  gemm(xb, D_, mlp_w1, D_, mlp_b1, nullptr, nullptr, hb, NX_, D_, D_, 1);
  gemm(hb, D_, mlp_w2, D_, mlp_b2, nullptr, (float*)d_out, nullptr, NX_, D_, 4, 0);
}
